// SimpleRouter_807453852023
// MI455X (gfx1250) — matrix operand order repaired and checked by reading, not compiled, not run on hardware
//
#include <hip/hip_runtime.h>
#include <hip/hip_bf16.h>
#include <math.h>

typedef __attribute__((ext_vector_type(16))) __bf16          v16bf;
typedef __attribute__((ext_vector_type(2)))  __bf16          v2bf;
typedef __attribute__((ext_vector_type(8)))  float           v8f;
typedef __attribute__((ext_vector_type(16))) unsigned short  v16u16;
typedef __attribute__((ext_vector_type(8)))  unsigned short  v8u16;

#define HIDDEN 4096
#define NEXP   256
#define KCHUNK 1024                 // K elems staged per chunk
#define LROW   (KCHUNK + 8)         // padded LDS row stride (bf16) -> conflict-free ds_load_b128
#define TOPK   8

__device__ __forceinline__ unsigned short f32_to_bf16(float f) {
    unsigned u = __builtin_bit_cast(unsigned, f);
    u += 0x7fffu + ((u >> 16) & 1u);            // round-to-nearest-even
    return (unsigned short)(u >> 16);
}

__device__ __forceinline__ unsigned pack_bf16x2(float a, float b) {
#if __has_builtin(__builtin_amdgcn_cvt_pk_bf16_f32)
    // single v_cvt_pk_bf16_f32
    return __builtin_bit_cast(unsigned, __builtin_amdgcn_cvt_pk_bf16_f32(a, b));
#else
    return (unsigned)f32_to_bf16(a) | ((unsigned)f32_to_bf16(b) << 16);
#endif
}

// -------- pre-pass: W f32 -> bf16 into workspace (2 MB, L2-resident) --------
__global__ __launch_bounds__(256)
void convert_w_bf16(const float* __restrict__ W, unsigned int* __restrict__ Wb2) {
    int q = blockIdx.x * 256 + threadIdx.x;        // quad index (4 f32 each)
    float4 f = ((const float4*)W)[q];
    uint2 o;
    o.x = pack_bf16x2(f.x, f.y);
    o.y = pack_bf16x2(f.z, f.w);
    ((uint2*)Wb2)[q] = o;
}

// -------- fused gate GEMM (bf16 WMMA) + sigmoid + top-8 + normalize --------
template <bool PRECONV>
__global__ __launch_bounds__(512)
void router_gate_topk(const float* __restrict__ x,
                      const float* __restrict__ W,
                      const unsigned short* __restrict__ Wb,
                      const float* __restrict__ bias,
                      int*   __restrict__ out_idx,
                      float* __restrict__ out_w,
                      int tokens)
{
    // 16 rows x LROW bf16 (~32.3 KB); later reused as 16x256 f32 logits tile.
    __shared__ unsigned int smem[(16 * LROW) / 2];
    unsigned short* sbf  = (unsigned short*)smem;
    float*          slog = (float*)smem;

    const int tid    = threadIdx.x;
    const int wave   = tid >> 5;        // 0..15 -> expert tile
    const int lane   = tid & 31;
    const int m0     = blockIdx.x * 16; // 16 tokens per block
    if (m0 >= tokens) return;

    const int nLane  = lane & 15;
    const int laneHi = lane >> 4;
    const int expert = wave * 16 + nLane;

    const float b = bias[expert];
    v8f acc;
#pragma unroll
    for (int i = 0; i < 8; ++i) acc[i] = b;

    const float*          xbase = x + (size_t)m0 * HIDDEN;
    const float*          wrow  = W  + (size_t)expert * HIDDEN;
    const unsigned short* wrowb = PRECONV ? (Wb + (size_t)expert * HIDDEN) : nullptr;

    for (int kc = 0; kc < HIDDEN; kc += KCHUNK) {
        // ---- stage x[m0..m0+15][kc..kc+KCHUNK) -> LDS bf16 (float4 loads) ----
        // 16*KCHUNK/4 = 4096 quads over 512 threads -> 8 quads each, coalesced.
#pragma unroll
        for (int i = 0; i < (16 * KCHUNK / 4) / 512; ++i) {
            int p   = tid + i * 512;
            int row = p >> 8;                       // KCHUNK/4 = 256 quads/row
            int cq  = p & 255;
            const float* gp = xbase + (size_t)row * HIDDEN + kc + cq * 4;
            float4 f = *(const float4*)gp;
            if (kc + KCHUNK < HIDDEN)               // warm L2 for next chunk
                __builtin_prefetch(gp + KCHUNK, 0, 3);
            uint2 o;
            o.x = pack_bf16x2(f.x, f.y);
            o.y = pack_bf16x2(f.z, f.w);
            *(uint2*)(smem + row * (LROW / 2) + cq * 2) = o;
        }
        __syncthreads();

        // ---- WMMA over this K chunk, step 32 ----
        for (int kb = 0; kb < KCHUNK; kb += 32) {
            // A fragment (16x32 bf16) from LDS (padded rows -> conflict-free):
            // lanes 0-15: K 0..15; lanes 16-31: K 16..31
            const int koffA = laneHi * 16;
            const unsigned short* pa = sbf + nLane * LROW + kb + koffA;
            v8u16 alo = *(const v8u16*)(pa);
            v8u16 ahi = *(const v8u16*)(pa + 8);
            v16u16 abits;
#pragma unroll
            for (int i = 0; i < 8; ++i) { abits[i] = alo[i]; abits[i + 8] = ahi[i]; }

            // B fragment (32x16 bf16): lane holds 16 contiguous K for column
            // N = lane&15; lanes 0-15 -> K 0..15, lanes 16-31 -> K 16..31.
            const int koffB = laneHi * 16;
            v16u16 bbits;
            if (PRECONV) {
                bbits = *(const v16u16*)(wrowb + kc + kb + koffB);  // 2x b128, no VALU
            } else {
                const float4* pw = (const float4*)(wrow + kc + kb + koffB);
#pragma unroll
                for (int q = 0; q < 4; ++q) {
                    float4 f = pw[q];
                    bbits[q * 4 + 0] = f32_to_bf16(f.x);
                    bbits[q * 4 + 1] = f32_to_bf16(f.y);
                    bbits[q * 4 + 2] = f32_to_bf16(f.z);
                    bbits[q * 4 + 3] = f32_to_bf16(f.w);
                }
            }

            v16bf av = __builtin_bit_cast(v16bf, abits);
            v16bf bv = __builtin_bit_cast(v16bf, bbits);
            acc = __builtin_amdgcn_wmma_f32_16x16x32_bf16(
                      false, av, false, bv, (short)0, acc, false, false);
        }
        __syncthreads();
    }

    // ---- spill logits tile to LDS as [16 tokens][256 experts] f32 ----
#pragma unroll
    for (int i = 0; i < 8; ++i) {
        int M = i + laneHi * 8;                 // C/D layout: VGPR i -> row M
        slog[M * NEXP + wave * 16 + nLane] = acc[i];
    }
    __syncthreads();

    // ---- fused sigmoid + top-8 + normalize: wave w handles token m0+w ----
    {
        const int t = m0 + wave;
        float v[8];
#pragma unroll
        for (int j = 0; j < 8; ++j) {
            float lg = slog[wave * NEXP + lane * 8 + j];
            // fast sigmoid: v_exp_f32 + v_rcp_f32
            v[j] = __builtin_amdgcn_rcpf(1.0f + __expf(-lg));
        }

        float topv[TOPK];
        int   topi[TOPK];
#pragma unroll
        for (int s = 0; s < TOPK; ++s) {
            float bv2 = v[0]; int bj = 0;
#pragma unroll
            for (int j = 1; j < 8; ++j)
                if (v[j] > bv2) { bv2 = v[j]; bj = j; }
            int bidx = lane * 8 + bj;
#pragma unroll
            for (int off = 16; off > 0; off >>= 1) {
                float ov = __shfl_xor(bv2, off, 32);
                int   oi = __shfl_xor(bidx, off, 32);
                if (ov > bv2 || (ov == bv2 && oi < bidx)) { bv2 = ov; bidx = oi; }
            }
            topv[s] = bv2;
            topi[s] = bidx;
            if ((bidx >> 3) == lane) v[bidx & 7] = -INFINITY;
        }

        float sum = 1e-8f;
#pragma unroll
        for (int s = 0; s < TOPK; ++s) sum += topv[s];
        float rs = __builtin_amdgcn_rcpf(sum);

        if (lane < TOPK) {
            out_idx[(size_t)t * TOPK + lane] = topi[lane];
            out_w  [(size_t)t * TOPK + lane] = topv[lane] * rs;
        }
    }
}

extern "C" void kernel_launch(void* const* d_in, const int* in_sizes, int n_in,
                              void* d_out, int out_size, void* d_ws, size_t ws_size,
                              hipStream_t stream) {
    (void)n_in; (void)out_size;
    const float* x    = (const float*)d_in[0];
    const float* W    = (const float*)d_in[1];
    const float* bias = (const float*)d_in[2];

    const int tokens = in_sizes[0] / HIDDEN;   // 16384

    // d_out layout: [tokens*8] int32 indices, then [tokens*8] f32 weights
    int*   out_idx = (int*)d_out;
    float* out_w   = (float*)d_out + (size_t)tokens * TOPK;

    dim3 grid(tokens / 16);
    const size_t wbBytes = (size_t)NEXP * HIDDEN * sizeof(unsigned short); // 2 MB

    if (ws_size >= wbBytes) {
        // pre-convert W to bf16 once (2 MB, stays in L2 for all 1024 tiles)
        convert_w_bf16<<<(NEXP * HIDDEN / 4) / 256, 256, 0, stream>>>(
            W, (unsigned int*)d_ws);
        router_gate_topk<true><<<grid, 512, 0, stream>>>(
            x, W, (const unsigned short*)d_ws, bias, out_idx, out_w, tokens);
    } else {
        router_gate_topk<false><<<grid, 512, 0, stream>>>(
            x, W, nullptr, bias, out_idx, out_w, tokens);
    }
}
